// TransformerCritic_73624329388134
// MI455X (gfx1250) — matrix operand order repaired and checked by reading, not compiled, not run on hardware
//
#include <hip/hip_runtime.h>
#include <hip/hip_bf16.h>
#include <math.h>

#define BATCH 4096
#define NTOK  64
#define OBS   192
#define ACTD  64
#define INDIM 256
#define DEMB  128
#define FH    64
#define FO    64

// Padded LDS leading dimensions (in halfs / floats): +8 halfs (+4 floats) so the
// per-row bank advance is 4 dwords -> 16 rows spread over all 64 LDS banks.
#define SA_LD  264
#define E_LD   136
#define QK_LD  136
#define VT_LD  72
#define NF_LD  264
#define W16_LD 72
#define H_LD   72
#define SC_LD  68

// Shared byte arena with Q/K overlaid by W16/H (disjoint lifetimes).
#define SMEM_SA   0
#define SMEM_E    (SMEM_SA + NTOK*SA_LD*2)    // 33792
#define SMEM_QW   (SMEM_E  + NTOK*E_LD*2)     // 51200: Q | W16
#define SMEM_H    (SMEM_QW + NTOK*W16_LD*2)   // 60416 (inside Q/K region)
#define SMEM_K    (SMEM_QW + NTOK*QK_LD*2)    // 68608
#define SMEM_VT   (SMEM_K  + NTOK*QK_LD*2)    // 86016
#define SMEM_NF   (SMEM_VT + DEMB*VT_LD*2)    // 104448
#define SMEM_SC   (SMEM_NF + NTOK*NF_LD*2)    // 138240
#define SMEM_SIZE (SMEM_SC + NTOK*SC_LD*4)    // 155648 B = 152 KB -> 2 blocks/WGP

typedef __attribute__((ext_vector_type(16))) _Float16 v16h;
typedef __attribute__((ext_vector_type(8)))  _Float16 v8h;
typedef __attribute__((ext_vector_type(4)))  _Float16 v4h;
typedef __attribute__((ext_vector_type(8)))  float    v8f;

// f16 weight offsets inside d_ws (in halfs)
#define OFF_WE   0
#define OFF_WQ   (128*256)
#define OFF_WK   (OFF_WQ + 128*128)
#define OFF_WV   (OFF_WK + 128*128)
#define OFF_WEQ  (OFF_WV + 128*128)
#define OFF_WF1  (OFF_WEQ + 128*256)
#define OFF_WF2  (OFF_WF1 + 64*256)

__global__ void cvt_f32_to_f16(const float* __restrict__ src, _Float16* __restrict__ dst, int n) {
    int i = blockIdx.x * blockDim.x + threadIdx.x;
    if (i < n) dst[i] = (_Float16)src[i];
}

// A fragment (16x32 f16): lane L holds row M=L&15; K = kb..kb+7 and kb+16..kb+23, kb=(L>>4)*8
__device__ __forceinline__ v16h load_fragA(const _Float16* A, int ld, int m0, int k0, int lane) {
    const _Float16* p = A + (m0 + (lane & 15)) * ld + k0 + ((lane >> 4) << 3);
    v8h lo = *(const v8h*)(p);
    v8h hi = *(const v8h*)(p + 16);
    v16h a;
#pragma unroll
    for (int i = 0; i < 8; ++i) { a[i] = lo[i]; a[i + 8] = hi[i]; }
    return a;
}

// B fragment (32x16 f16, K x N): lane L holds column N=L&15; K = kb..kb+7 and kb+16..kb+23, kb=k0+(L>>4)*8.
// For C = A * W^T with W row-major [N,K]: B[k][n] = W[n][k] -> contiguous row chunk of W.
__device__ __forceinline__ v16h load_fragB(const _Float16* W, int ld, int n0, int k0, int lane) {
    const _Float16* p = W + (n0 + (lane & 15)) * ld + k0 + ((lane >> 4) << 3);
    v8h lo = *(const v8h*)(p);
    v8h hi = *(const v8h*)(p + 16);
    v16h b;
#pragma unroll
    for (int i = 0; i < 8; ++i) { b[i] = lo[i]; b[i + 8] = hi[i]; }
    return b;
}

__device__ __forceinline__ v8f wmma_acc(v16h a, v16h b, v8f c) {
    return __builtin_amdgcn_wmma_f32_16x16x32_f16(false, a, false, b, (short)0, c, false, false);
}

__device__ __forceinline__ float lrelu(float x) { return fmaxf(x, 0.01f * x); }

__global__ __launch_bounds__(256, 1)
void critic_kernel(const float* __restrict__ states, const float* __restrict__ actions,
                   const float* __restrict__ b_embed, const float* __restrict__ b_embed_q,
                   const _Float16* __restrict__ wts, float* __restrict__ out)
{
    __shared__ __align__(16) unsigned char smem[SMEM_SIZE];
    _Float16* s_sa  = (_Float16*)(smem + SMEM_SA);
    _Float16* s_e   = (_Float16*)(smem + SMEM_E);
    _Float16* s_q   = (_Float16*)(smem + SMEM_QW);
    _Float16* s_w16 = (_Float16*)(smem + SMEM_QW);
    _Float16* s_h   = (_Float16*)(smem + SMEM_H);
    _Float16* s_k   = (_Float16*)(smem + SMEM_K);
    _Float16* s_vT  = (_Float16*)(smem + SMEM_VT);
    _Float16* s_nf  = (_Float16*)(smem + SMEM_NF);
    float*    s_sc  = (float*)   (smem + SMEM_SC);

    const int b    = blockIdx.x;
    const int tid  = threadIdx.x;
    const int lane = tid & 31;
    const int wave = tid >> 5;
    const int cn   = lane & 15;         // C-fragment column within tile
    const int mh   = (lane >> 4) << 3;  // C-fragment row offset (0 or 8)

    // ---- Stage 1: load sa = concat(states, actions) into LDS as f16 ----
    {
        const float4* st = (const float4*)(states + (size_t)b * NTOK * OBS);
#pragma unroll
        for (int i = 0; i < 12; ++i) {                 // 3072 float4
            int idx = tid + 256 * i;
            int n = idx / 48, f4 = idx % 48;
            float4 v = st[idx];
            v4h h; h[0] = (_Float16)v.x; h[1] = (_Float16)v.y; h[2] = (_Float16)v.z; h[3] = (_Float16)v.w;
            *(v4h*)(s_sa + n * SA_LD + f4 * 4) = h;
        }
        const float4* ac = (const float4*)(actions + (size_t)b * NTOK * ACTD);
#pragma unroll
        for (int i = 0; i < 4; ++i) {                  // 1024 float4
            int idx = tid + 256 * i;
            int n = idx / 16, f4 = idx % 16;
            float4 v = ac[idx];
            v4h h; h[0] = (_Float16)v.x; h[1] = (_Float16)v.y; h[2] = (_Float16)v.z; h[3] = (_Float16)v.w;
            *(v4h*)(s_sa + n * SA_LD + OBS + f4 * 4) = h;
        }
    }
    __syncthreads();

    // ---- Stage 2: e = lrelu(sa We^T + b), eq = lrelu(sa Weq^T + bq).
    //      16 macro-tiles (32x32 = 2x2 WMMA tiles), 2 per wave, K=256.  1 frag load / WMMA. ----
    for (int it = 0; it < 2; ++it) {
        int mt = wave + 8 * it;
        int which = mt >> 3;                // 0 = e, 1 = eq
        int sub = mt & 7;
        int m0 = (sub >> 2) << 5, n0 = (sub & 3) << 5;
        const _Float16* W = wts + (which ? OFF_WEQ : OFF_WE);
        const float* bias = which ? b_embed_q : b_embed;
        v8f a00 = {}, a01 = {}, a10 = {}, a11 = {};
#pragma unroll
        for (int k0 = 0; k0 < INDIM; k0 += 32) {
            v16h fa0 = load_fragA(s_sa, SA_LD, m0,      k0, lane);
            v16h fa1 = load_fragA(s_sa, SA_LD, m0 + 16, k0, lane);
            v16h fb0 = load_fragB(W, INDIM, n0,      k0, lane);
            v16h fb1 = load_fragB(W, INDIM, n0 + 16, k0, lane);
            a00 = wmma_acc(fa0, fb0, a00);
            a01 = wmma_acc(fa0, fb1, a01);
            a10 = wmma_acc(fa1, fb0, a10);
            a11 = wmma_acc(fa1, fb1, a11);
        }
        float bv0 = bias[n0 + cn], bv1 = bias[n0 + 16 + cn];
        _Float16* dst = which ? s_nf : s_e;
        int ldd = which ? NF_LD : E_LD;
#pragma unroll
        for (int i = 0; i < 8; ++i) {
            int r0 = m0 + i + mh, r1 = m0 + 16 + i + mh;
            dst[r0 * ldd + n0 + cn]      = (_Float16)lrelu(a00[i] + bv0);
            dst[r0 * ldd + n0 + 16 + cn] = (_Float16)lrelu(a01[i] + bv1);
            dst[r1 * ldd + n0 + cn]      = (_Float16)lrelu(a10[i] + bv0);
            dst[r1 * ldd + n0 + 16 + cn] = (_Float16)lrelu(a11[i] + bv1);
        }
    }
    __syncthreads();

    // ---- Stage 3: q,k,v = e W^T (v stored transposed).  24 macro-tiles (32x32), 3 per wave, K=128 ----
    for (int it = 0; it < 3; ++it) {
        int mt = wave + 8 * it;
        int which = mt >> 3;                // 0=q, 1=k, 2=v
        int sub = mt & 7;
        int m0 = (sub >> 2) << 5, n0 = (sub & 3) << 5;
        const _Float16* W = wts + (which == 0 ? OFF_WQ : which == 1 ? OFF_WK : OFF_WV);
        v8f a00 = {}, a01 = {}, a10 = {}, a11 = {};
#pragma unroll
        for (int k0 = 0; k0 < DEMB; k0 += 32) {
            v16h fa0 = load_fragA(s_e, E_LD, m0,      k0, lane);
            v16h fa1 = load_fragA(s_e, E_LD, m0 + 16, k0, lane);
            v16h fb0 = load_fragB(W, DEMB, n0,      k0, lane);
            v16h fb1 = load_fragB(W, DEMB, n0 + 16, k0, lane);
            a00 = wmma_acc(fa0, fb0, a00);
            a01 = wmma_acc(fa0, fb1, a01);
            a10 = wmma_acc(fa1, fb0, a10);
            a11 = wmma_acc(fa1, fb1, a11);
        }
        if (which == 2) {
            v8h h00, h01, h10, h11;          // 8 consecutive m -> one ds_store_b128 each
#pragma unroll
            for (int i = 0; i < 8; ++i) {
                h00[i] = (_Float16)a00[i]; h01[i] = (_Float16)a01[i];
                h10[i] = (_Float16)a10[i]; h11[i] = (_Float16)a11[i];
            }
            *(v8h*)(s_vT + (n0 + cn)      * VT_LD + m0 + mh)      = h00;
            *(v8h*)(s_vT + (n0 + 16 + cn) * VT_LD + m0 + mh)      = h01;
            *(v8h*)(s_vT + (n0 + cn)      * VT_LD + m0 + 16 + mh) = h10;
            *(v8h*)(s_vT + (n0 + 16 + cn) * VT_LD + m0 + 16 + mh) = h11;
        } else {
            _Float16* dst = (which == 1) ? s_k : s_q;
#pragma unroll
            for (int i = 0; i < 8; ++i) {
                int r0 = m0 + i + mh, r1 = m0 + 16 + i + mh;
                dst[r0 * QK_LD + n0 + cn]      = (_Float16)a00[i];
                dst[r0 * QK_LD + n0 + 16 + cn] = (_Float16)a01[i];
                dst[r1 * QK_LD + n0 + cn]      = (_Float16)a10[i];
                dst[r1 * QK_LD + n0 + 16 + cn] = (_Float16)a11[i];
            }
        }
    }
    __syncthreads();

    // ---- Stage 4: scores = q k^T / sqrt(D).  8 macro-tiles (16x32), 1 per wave, K=128 ----
    const float scale = 0.08838834764831845f;   // 1/sqrt(128)
    {
        int m0 = (wave >> 1) << 4, n0 = (wave & 1) << 5;
        v8f a0 = {}, a1 = {};
#pragma unroll
        for (int k0 = 0; k0 < DEMB; k0 += 32) {
            v16h fa = load_fragA(s_q, QK_LD, m0, k0, lane);
            v16h fb0 = load_fragB(s_k, QK_LD, n0,      k0, lane);
            v16h fb1 = load_fragB(s_k, QK_LD, n0 + 16, k0, lane);
            a0 = wmma_acc(fa, fb0, a0);
            a1 = wmma_acc(fa, fb1, a1);
        }
#pragma unroll
        for (int i = 0; i < 8; ++i) {
            int r = m0 + i + mh;
            s_sc[r * SC_LD + n0 + cn]      = a0[i] * scale;
            s_sc[r * SC_LD + n0 + 16 + cn] = a1[i] * scale;
        }
    }
    __syncthreads();

    // ---- Stage 5: row softmax, 4 threads/row, cross-lane reduce via shfl_xor ----
    {
        int r = tid >> 2, q4 = tid & 3;
        float* row = s_sc + r * SC_LD + q4 * 16;
        float vb[16];
#pragma unroll
        for (int i = 0; i < 4; ++i) *(float4*)(vb + 4 * i) = *(const float4*)(row + 4 * i);
        float mx = vb[0];
#pragma unroll
        for (int j = 1; j < 16; ++j) mx = fmaxf(mx, vb[j]);
        mx = fmaxf(mx, __shfl_xor(mx, 1, 32));
        mx = fmaxf(mx, __shfl_xor(mx, 2, 32));
        float sum = 0.f;
#pragma unroll
        for (int j = 0; j < 16; ++j) { vb[j] = __expf(vb[j] - mx); sum += vb[j]; }
        sum += __shfl_xor(sum, 1, 32);
        sum += __shfl_xor(sum, 2, 32);
        float inv = 1.f / sum;
#pragma unroll
        for (int j = 0; j < 16; ++j) vb[j] *= inv;
#pragma unroll
        for (int i = 0; i < 4; ++i) *(float4*)(row + 4 * i) = *(const float4*)(vb + 4 * i);
        v8h h0, h1;
#pragma unroll
        for (int j = 0; j < 8; ++j) { h0[j] = (_Float16)vb[j]; h1[j] = (_Float16)vb[j + 8]; }
        *(v8h*)(s_w16 + r * W16_LD + q4 * 16)     = h0;
        *(v8h*)(s_w16 + r * W16_LD + q4 * 16 + 8) = h1;
    }
    __syncthreads();

    // ---- Stage 6a: write attention weights (output #2, f32, coalesced) ----
    {
        float* wout = out + (size_t)BATCH * NTOK * FO + (size_t)b * NTOK * NTOK;
#pragma unroll
        for (int i = 0; i < 4; ++i) {
            int idx4 = tid + 256 * i;                  // 1024 float4
            int rr = idx4 >> 4, c4 = idx4 & 15;
            *(float4*)(wout + rr * NTOK + c4 * 4) = *(const float4*)(s_sc + rr * SC_LD + c4 * 4);
        }
    }
    // ---- Stage 6b: x = weight @ v.  8 macro-tiles (32x32), 1 per wave, K=64 ----
    {
        int m0 = (wave >> 2) << 5, n0 = (wave & 3) << 5;
        v8f a00 = {}, a01 = {}, a10 = {}, a11 = {};
#pragma unroll
        for (int k0 = 0; k0 < NTOK; k0 += 32) {
            v16h fa0 = load_fragA(s_w16, W16_LD, m0,      k0, lane);
            v16h fa1 = load_fragA(s_w16, W16_LD, m0 + 16, k0, lane);
            v16h fb0 = load_fragB(s_vT, VT_LD, n0,      k0, lane);
            v16h fb1 = load_fragB(s_vT, VT_LD, n0 + 16, k0, lane);
            a00 = wmma_acc(fa0, fb0, a00);
            a01 = wmma_acc(fa0, fb1, a01);
            a10 = wmma_acc(fa1, fb0, a10);
            a11 = wmma_acc(fa1, fb1, a11);
        }
#pragma unroll
        for (int i = 0; i < 8; ++i) {
            int r0 = m0 + i + mh, r1 = m0 + 16 + i + mh;
            s_nf[r0 * NF_LD + DEMB + n0 + cn]      = (_Float16)a00[i];
            s_nf[r0 * NF_LD + DEMB + n0 + 16 + cn] = (_Float16)a01[i];
            s_nf[r1 * NF_LD + DEMB + n0 + cn]      = (_Float16)a10[i];
            s_nf[r1 * NF_LD + DEMB + n0 + 16 + cn] = (_Float16)a11[i];
        }
    }
    __syncthreads();

    // ---- Stage 7: h = lrelu(nf @ Wf1^T).  8 macro-tiles (16x32), 1 per wave, K=256 ----
    {
        int m0 = (wave >> 1) << 4, n0 = (wave & 1) << 5;
        v8f a0 = {}, a1 = {};
#pragma unroll
        for (int k0 = 0; k0 < 2 * DEMB; k0 += 32) {
            v16h fa = load_fragA(s_nf, NF_LD, m0, k0, lane);
            v16h fb0 = load_fragB(wts + OFF_WF1, 2 * DEMB, n0,      k0, lane);
            v16h fb1 = load_fragB(wts + OFF_WF1, 2 * DEMB, n0 + 16, k0, lane);
            a0 = wmma_acc(fa, fb0, a0);
            a1 = wmma_acc(fa, fb1, a1);
        }
#pragma unroll
        for (int i = 0; i < 8; ++i) {
            int r = m0 + i + mh;
            s_h[r * H_LD + n0 + cn]      = (_Float16)lrelu(a0[i]);
            s_h[r * H_LD + n0 + 16 + cn] = (_Float16)lrelu(a1[i]);
        }
    }
    __syncthreads();

    // ---- Stage 8: Value = h @ Wf2^T (output #1, f32).  8 macro-tiles (16x32), 1 per wave, K=64 ----
    {
        float* vout = out + (size_t)b * NTOK * FO;
        int m0 = (wave >> 1) << 4, n0 = (wave & 1) << 5;
        v8f a0 = {}, a1 = {};
#pragma unroll
        for (int k0 = 0; k0 < FH; k0 += 32) {
            v16h fa = load_fragA(s_h, H_LD, m0, k0, lane);
            v16h fb0 = load_fragB(wts + OFF_WF2, FH, n0,      k0, lane);
            v16h fb1 = load_fragB(wts + OFF_WF2, FH, n0 + 16, k0, lane);
            a0 = wmma_acc(fa, fb0, a0);
            a1 = wmma_acc(fa, fb1, a1);
        }
#pragma unroll
        for (int i = 0; i < 8; ++i) {
            int r = m0 + i + mh;
            vout[r * FO + n0 + cn]      = a0[i];
            vout[r * FO + n0 + 16 + cn] = a1[i];
        }
    }
}

extern "C" void kernel_launch(void* const* d_in, const int* in_sizes, int n_in,
                              void* d_out, int out_size, void* d_ws, size_t ws_size,
                              hipStream_t stream) {
    (void)in_sizes; (void)n_in; (void)out_size; (void)ws_size;
    const float* states    = (const float*)d_in[0];
    const float* actions   = (const float*)d_in[1];
    const float* W_embed   = (const float*)d_in[2];
    const float* b_embed   = (const float*)d_in[3];
    const float* W_k       = (const float*)d_in[4];
    const float* W_q       = (const float*)d_in[5];
    const float* W_v       = (const float*)d_in[6];
    const float* W_embed_q = (const float*)d_in[7];
    const float* b_embed_q = (const float*)d_in[8];
    const float* W_f1      = (const float*)d_in[9];
    const float* W_f2      = (const float*)d_in[10];
    _Float16* ws = (_Float16*)d_ws;

    auto cvt = [&](const float* s, int off, int n) {
        cvt_f32_to_f16<<<(n + 255) / 256, 256, 0, stream>>>(s, ws + off, n);
    };
    cvt(W_embed,   OFF_WE,  128 * 256);
    cvt(W_q,       OFF_WQ,  128 * 128);
    cvt(W_k,       OFF_WK,  128 * 128);
    cvt(W_v,       OFF_WV,  128 * 128);
    cvt(W_embed_q, OFF_WEQ, 128 * 256);
    cvt(W_f1,      OFF_WF1, 64 * 256);
    cvt(W_f2,      OFF_WF2, 64 * 64);

    critic_kernel<<<BATCH, 256, 0, stream>>>(states, actions, b_embed, b_embed_q, ws, (float*)d_out);
}
